// HadamardQuantizedLoraLinear_39986145526403
// MI455X (gfx1250) — 8-bit matrix operand order repaired and checked by reading, compiled, not run on hardware
//
#include <hip/hip_runtime.h>
#include <hip/hip_bf16.h>


typedef int v8i __attribute__((ext_vector_type(8)));

#define D_IN   4096
#define ROWS_X 8192
#define ROWS_W 4096
#define R_LORA 16

#if defined(__has_builtin)
#  if __has_builtin(__builtin_amdgcn_global_load_async_to_lds_b128)
#    define HAVE_ASYNC_LDS 1
#  endif
#endif

#ifdef HAVE_ASYNC_LDS
typedef int v4i_vs __attribute__((vector_size(16)));
typedef v4i_vs __attribute__((address_space(1)))* gv4i_ptr;
typedef v4i_vs __attribute__((address_space(3)))* lv4i_ptr;

template <int OFF>
__device__ __forceinline__ void async_copy16(const signed char* g, signed char* l) {
    __builtin_amdgcn_global_load_async_to_lds_b128((gv4i_ptr)g, (lv4i_ptr)l, OFF, 0);
}
template <int N>
__device__ __forceinline__ void async_wait() {
#if __has_builtin(__builtin_amdgcn_s_wait_asynccnt)
    __builtin_amdgcn_s_wait_asynccnt(N);
#else
    asm volatile("s_wait_asynccnt %0" :: "i"(N) : "memory");
#endif
}
#endif

__global__ __launch_bounds__(256) void fwht_quant_kernel(
    const float* __restrict__ in, signed char* __restrict__ outq,
    float* __restrict__ scale)
{
    __shared__ float buf[D_IN];
    __shared__ float wred[8];
    __shared__ float sshared;
    const int tid = threadIdx.x;
    const size_t row = blockIdx.x;
    const float* src = in + row * D_IN;

    for (int i = tid; i < D_IN; i += 256) buf[i] = src[i] * 0.015625f;
    __syncthreads();

    for (int h = 1; h < D_IN; h <<= 1) {
        for (int p = tid; p < D_IN / 2; p += 256) {
            const int i = ((p & ~(h - 1)) << 1) | (p & (h - 1));
            const int j = i + h;
            const float a = buf[i], b = buf[j];
            buf[i] = a + b;
            buf[j] = a - b;
        }
        __syncthreads();
    }

    float mx = 0.f;
    for (int i = tid; i < D_IN; i += 256) mx = fmaxf(mx, fabsf(buf[i]));
    for (int off = 16; off; off >>= 1) mx = fmaxf(mx, __shfl_xor(mx, off, 32));
    if ((tid & 31) == 0) wred[tid >> 5] = mx;
    __syncthreads();
    if (tid == 0) {
        float m = wred[0];
        for (int i = 1; i < 8; ++i) m = fmaxf(m, wred[i]);
        const float s = fmaxf(m * (1.0f / 127.0f), 1e-8f);
        sshared = s;
        scale[row] = s;
    }
    __syncthreads();

    const float invs = 1.0f / sshared;
    signed char* dq = outq + row * D_IN;
    for (int i = tid; i < D_IN; i += 256) {
        float q = rintf(buf[i] * invs);
        q = fminf(fmaxf(q, -127.f), 127.f);
        dq[i] = (signed char)q;
    }
}

__global__ __launch_bounds__(256) void lora_t_kernel(
    const float* __restrict__ x, const float* __restrict__ loraA,
    float* __restrict__ t)
{
    const int wave = threadIdx.x >> 5;
    const int lane = threadIdx.x & 31;
    const size_t row = (size_t)blockIdx.x * 8 + wave;
    const float* xr = x + row * D_IN;

    float acc[R_LORA];
#pragma unroll
    for (int r = 0; r < R_LORA; ++r) acc[r] = 0.f;

    for (int k = lane * 4; k < D_IN; k += 32 * 4) {
        const float4 xv = *(const float4*)(xr + k);
#pragma unroll
        for (int r = 0; r < R_LORA; ++r) {
            const float4 av = *(const float4*)(loraA + r * D_IN + k);
            acc[r] += xv.x * av.x + xv.y * av.y + xv.z * av.z + xv.w * av.w;
        }
    }
#pragma unroll
    for (int r = 0; r < R_LORA; ++r)
        for (int off = 16; off; off >>= 1)
            acc[r] += __shfl_xor(acc[r], off, 32);
    if (lane == 0) {
#pragma unroll
        for (int r = 0; r < R_LORA; ++r) t[row * R_LORA + r] = acc[r];
    }
}

__global__ __launch_bounds__(256) void gemm_iu8_kernel(
    const signed char* __restrict__ qx, const signed char* __restrict__ qw,
    const float* __restrict__ sx, const float* __restrict__ sw,
    const float* __restrict__ t,  const float* __restrict__ loraB,
    float* __restrict__ out)
{
    constexpr int LDT  = 80;
    constexpr int TILE = 128 * LDT;
#ifdef HAVE_ASYNC_LDS
    constexpr int NBUF = 2;
#else
    constexpr int NBUF = 1;
#endif
    __shared__ signed char As[NBUF * TILE];
    __shared__ signed char Bs[NBUF * TILE];
    __shared__ float Ts[128 * R_LORA];
    __shared__ float Ls[128 * R_LORA];
    __shared__ float Sx[128];
    __shared__ float Sw[128];

    const int tid  = threadIdx.x;
    const int lane = tid & 31;
    const int wave = tid >> 5;
    const int wm   = wave >> 1;
    const int wn   = wave & 1;
    const int bm   = blockIdx.y * 128;
    const int bn   = blockIdx.x * 128;
    const int m16  = lane & 15;
    const int hi   = lane >> 4;

    for (int i = tid; i < 128 * R_LORA; i += 256) {
        Ts[i] = t[(size_t)(bm + (i >> 4)) * R_LORA + (i & 15)];
        Ls[i] = loraB[(size_t)(bn + (i >> 4)) * R_LORA + (i & 15)];
    }
    if (tid < 128) Sx[tid] = sx[bm + tid];
    else           Sw[tid - 128] = sw[bn + (tid - 128)];

    const int lrow = tid >> 1;
    const int lcol = (tid & 1) * 32;
    const signed char* gA = qx + (size_t)(bm + lrow) * D_IN + lcol;
    const signed char* gB = qw + (size_t)(bn + lrow) * D_IN + lcol;

    const v8i vzero = {0, 0, 0, 0, 0, 0, 0, 0};
    v8i acc[2][4];
#pragma unroll
    for (int a = 0; a < 2; ++a)
#pragma unroll
        for (int b = 0; b < 4; ++b) acc[a][b] = vzero;

    auto computeStep = [&](const signed char* baseA, const signed char* baseB) {
        v8i afr[2], bfr[4];
#pragma unroll
        for (int tm = 0; tm < 2; ++tm) {
            const int* ap = (const int*)(baseA + (wm * 32 + tm * 16 + m16) * LDT + hi * 16);
            v8i a;
            a[0] = ap[0];  a[1] = ap[1];
            a[2] = ap[2];  a[3] = ap[3];
            a[4] = ap[8];  a[5] = ap[9];
            a[6] = ap[10]; a[7] = ap[11];
            afr[tm] = a;
        }
#pragma unroll
        for (int tn = 0; tn < 4; ++tn) {
            const signed char* bb = baseB + (wn * 64 + tn * 16 + m16) * LDT + hi * 16;
            const int4 b0 = *(const int4*)(bb);
            const int4 b1 = *(const int4*)(bb + 32);
            v8i b;
            b[0] = b0.x; b[1] = b0.y; b[2] = b0.z; b[3] = b0.w;
            b[4] = b1.x; b[5] = b1.y; b[6] = b1.z; b[7] = b1.w;
            bfr[tn] = b;
        }
#pragma unroll
        for (int tm = 0; tm < 2; ++tm)
#pragma unroll
            for (int tn = 0; tn < 4; ++tn)
                acc[tm][tn] = __builtin_amdgcn_wmma_i32_16x16x64_iu8(
                    true, afr[tm], true, bfr[tn], acc[tm][tn], false, false);
    };

#ifdef HAVE_ASYNC_LDS
    auto issue = [&](int k, int buf) {
        signed char* la = As + buf * TILE + lrow * LDT + lcol;
        signed char* lb = Bs + buf * TILE + lrow * LDT + lcol;
        async_copy16<0>(gA + k,      la);
        async_copy16<0>(gA + k + 16, la + 16);
        async_copy16<0>(gB + k,      lb);
        async_copy16<0>(gB + k + 16, lb + 16);
    };
    issue(0, 0);
    for (int kt = 0; kt < D_IN; kt += 64) {
        const int cur = (kt >> 6) & 1;
        if (kt + 64 < D_IN) {
            issue(kt + 64, cur ^ 1);
            async_wait<4>();
        } else {
            async_wait<0>();
        }
        __syncthreads();
        computeStep(As + cur * TILE, Bs + cur * TILE);
        __syncthreads();
    }
#else
    int4 ra0, ra1, rb0, rb1;
    auto loadG = [&](int k) {
        ra0 = *(const int4*)(gA + k);
        ra1 = *(const int4*)(gA + k + 16);
        rb0 = *(const int4*)(gB + k);
        rb1 = *(const int4*)(gB + k + 16);
    };
    loadG(0);
    for (int kt = 0; kt < D_IN; kt += 64) {
        __syncthreads();
        *(int4*)(As + lrow * LDT + lcol)      = ra0;
        *(int4*)(As + lrow * LDT + lcol + 16) = ra1;
        *(int4*)(Bs + lrow * LDT + lcol)      = rb0;
        *(int4*)(Bs + lrow * LDT + lcol + 16) = rb1;
        __syncthreads();
        if (kt + 64 < D_IN) loadG(kt + 64);
        computeStep(As, Bs);
    }
#endif

#pragma unroll
    for (int tm = 0; tm < 2; ++tm) {
#pragma unroll
        for (int tn = 0; tn < 4; ++tn) {
            const int nl  = wn * 64 + tn * 16 + m16;
            const float swn = Sw[nl];
            const float4 L0 = *(const float4*)(Ls + nl * 16 + 0);
            const float4 L1 = *(const float4*)(Ls + nl * 16 + 4);
            const float4 L2 = *(const float4*)(Ls + nl * 16 + 8);
            const float4 L3 = *(const float4*)(Ls + nl * 16 + 12);
#pragma unroll
            for (int j = 0; j < 8; ++j) {
                const int ml = wm * 32 + tm * 16 + hi * 8 + j;
                const float4 T0 = *(const float4*)(Ts + ml * 16 + 0);
                const float4 T1 = *(const float4*)(Ts + ml * 16 + 4);
                const float4 T2 = *(const float4*)(Ts + ml * 16 + 8);
                const float4 T3 = *(const float4*)(Ts + ml * 16 + 12);
                float lora = T0.x * L0.x + T0.y * L0.y + T0.z * L0.z + T0.w * L0.w
                           + T1.x * L1.x + T1.y * L1.y + T1.z * L1.z + T1.w * L1.w
                           + T2.x * L2.x + T2.y * L2.y + T2.z * L2.z + T2.w * L2.w
                           + T3.x * L3.x + T3.y * L3.y + T3.z * L3.z + T3.w * L3.w;
                out[(size_t)(bm + ml) * (size_t)ROWS_W + (bn + nl)] =
                    (float)acc[tm][tn][j] * (Sx[ml] * swn) + 2.0f * lora;
            }
        }
    }
}

extern "C" void kernel_launch(void* const* d_in, const int* in_sizes, int n_in,
                              void* d_out, int out_size, void* d_ws, size_t ws_size,
                              hipStream_t stream) {
    (void)in_sizes; (void)n_in; (void)out_size; (void)ws_size;
    const float* x      = (const float*)d_in[0];
    const float* weight = (const float*)d_in[1];
    const float* lora_A = (const float*)d_in[2];
    const float* lora_B = (const float*)d_in[3];
    float* out = (float*)d_out;

    char* ws = (char*)d_ws;
    signed char* qx = (signed char*)(ws);
    signed char* qw = (signed char*)(ws + 33554432);
    float* sx = (float*)(ws + 50331648);
    float* sw = (float*)(ws + 50364416);
    float* t  = (float*)(ws + 50380800);

    fwht_quant_kernel<<<ROWS_X, 256, 0, stream>>>(x, qx, sx);
    fwht_quant_kernel<<<ROWS_W, 256, 0, stream>>>(weight, qw, sw);
    lora_t_kernel<<<ROWS_X / 8, 256, 0, stream>>>(x, lora_A, t);
    gemm_iu8_kernel<<<dim3(ROWS_W / 128, ROWS_X / 128), 256, 0, stream>>>(
        qx, qw, sx, sw, t, lora_B, out);
}
